// LearnedTokenizer_4836133175669
// MI455X (gfx1250) — matrix operand order repaired and checked by reading, not compiled, not run on hardware
//
#include <hip/hip_runtime.h>

// ---------------------------------------------------------------------------
// LearnedTokenizer on MI455X (gfx1250): bf16 WMMA pipeline
//   B=8, K=8192, C=512, T=256, H=8, Dh=64
// ---------------------------------------------------------------------------

typedef __bf16 bf16_t;
typedef __attribute__((ext_vector_type(16))) __bf16 v16bf;
typedef __attribute__((ext_vector_type(8)))  __bf16 v8bf;
typedef __attribute__((ext_vector_type(8)))  float  v8f;

#define BATCH 8
#define KLEN  8192
#define CDIM  512
#define TLEN  256
#define HEADS 8
#define DH    64

// -------- async global->LDS path (CDNA5), guarded ---------------------------
#if defined(__has_builtin)
#if __has_builtin(__builtin_amdgcn_global_load_async_to_lds_b128)
#define HAVE_ASYNC_LDS 1
#endif
#endif
#ifndef HAVE_ASYNC_LDS
#define HAVE_ASYNC_LDS 0
#endif

#if HAVE_ASYNC_LDS
typedef int v4i_t __attribute__((ext_vector_type(4)));
typedef __attribute__((address_space(1))) v4i_t* as1_v4i;
typedef __attribute__((address_space(3))) v4i_t* as3_v4i;
__device__ __forceinline__ void async_b128(const void* g, void* l) {
    __builtin_amdgcn_global_load_async_to_lds_b128(
        (as1_v4i)(v4i_t*)const_cast<void*>(g), (as3_v4i)(v4i_t*)l, 0, 0);
}
#define WAIT_ASYNC() asm volatile("s_wait_asynccnt 0" ::: "memory")
#endif

__device__ __forceinline__ v8f wmma_bf16(v16bf a, v16bf b, v8f c) {
    return __builtin_amdgcn_wmma_f32_16x16x32_bf16(false, a, false, b,
                                                   (short)0, c, false, false);
}

// A-operand fragment (16x32): row=lane%16, K cols {2*c0..2*c0+15} (contiguous)
// with c0 = 0 for lanes 0-15, c0 = 8 for lanes 16-31.
__device__ __forceinline__ v16bf ldA16(const bf16_t* rowp, int c0) {
    v8bf lo = *reinterpret_cast<const v8bf*>(rowp + 2 * c0);
    v8bf hi = *reinterpret_cast<const v8bf*>(rowp + 2 * c0 + 8);
    v16bf r;
#pragma unroll
    for (int i = 0; i < 8; ++i) { r[i] = lo[i]; r[i + 8] = hi[i]; }
    return r;
}

// B-operand fragment (32x16): lane = column, 16 consecutive K values
// (half-wave selects K 0..15 vs 16..31; caller bakes that into the pointer).
__device__ __forceinline__ v16bf ldB16(const bf16_t* p) {
    v8bf lo = *reinterpret_cast<const v8bf*>(p);
    v8bf hi = *reinterpret_cast<const v8bf*>(p + 8);
    v16bf r;
#pragma unroll
    for (int i = 0; i < 8; ++i) { r[i] = lo[i]; r[i + 8] = hi[i]; }
    return r;
}

// max-reduction over a 16-lane half via ds_swizzle (xor butterflies)
__device__ __forceinline__ float max_red16(float v) {
    v = fmaxf(v, __int_as_float(__builtin_amdgcn_ds_swizzle(__float_as_int(v), 0x041F)));
    v = fmaxf(v, __int_as_float(__builtin_amdgcn_ds_swizzle(__float_as_int(v), 0x081F)));
    v = fmaxf(v, __int_as_float(__builtin_amdgcn_ds_swizzle(__float_as_int(v), 0x101F)));
    v = fmaxf(v, __int_as_float(__builtin_amdgcn_ds_swizzle(__float_as_int(v), 0x201F)));
    return v;
}

// ---------------------------------------------------------------------------
// Kernel 1: projection GEMM  out = A(f32) @ W^T + bias  -> bf16 scratch
// MODE 0: q = queries@Wq^T * 1/sqrt(Dh), layout (H,T,Dh)
// MODE 1: k,v = x@[Wk;Wv]^T, k layout (B,H,K,Dh), v layout (B,H,Dh,K)
// Tile: 256(M) x 64(N) per 256-thread WG; each wave owns 32 rows (2 A-frags
// sharing 4 B-frags -> 8 WMMA per kstep); BK=32; f32->bf16 staged via LDS.
// ---------------------------------------------------------------------------
template <int MODE>
__global__ __launch_bounds__(256) void proj_kernel(
    const float* __restrict__ A, const float* __restrict__ W,
    const float* __restrict__ bias,
    bf16_t* __restrict__ q_s, bf16_t* __restrict__ k_s, bf16_t* __restrict__ v_t)
{
    __shared__ bf16_t XS[256][40];  // 40 = 32 + 8 pad (conflict-free strides)
    __shared__ bf16_t WS[64][40];

    const int tid   = threadIdx.x;
    const int wv    = tid >> 5;
    const int l     = tid & 31;
    const int nbase = blockIdx.x * 64;
    const int mbase = blockIdx.y * 256;

    v8f acc[2][4] = {};

    for (int kc = 0; kc < CDIM; kc += 32) {
        // stage A (256x32 f32 -> bf16)
#pragma unroll
        for (int p = 0; p < 8; ++p) {
            const int r = (tid >> 3) + 32 * p;
            const int c = (tid & 7) << 2;
            const float4 f = *reinterpret_cast<const float4*>(
                A + (size_t)(mbase + r) * CDIM + kc + c);
            XS[r][c + 0] = (bf16_t)f.x; XS[r][c + 1] = (bf16_t)f.y;
            XS[r][c + 2] = (bf16_t)f.z; XS[r][c + 3] = (bf16_t)f.w;
        }
        // stage W (64x32 f32 -> bf16)
#pragma unroll
        for (int p = 0; p < 2; ++p) {
            const int r = (tid >> 3) + 32 * p;
            const int c = (tid & 7) << 2;
            const float4 f = *reinterpret_cast<const float4*>(
                W + (size_t)(nbase + r) * CDIM + kc + c);
            WS[r][c + 0] = (bf16_t)f.x; WS[r][c + 1] = (bf16_t)f.y;
            WS[r][c + 2] = (bf16_t)f.z; WS[r][c + 3] = (bf16_t)f.w;
        }
        __syncthreads();

        const int rbase = wv * 32 + (l & 15);
        const int c0    = (l < 16) ? 0 : 8;
        const v16bf a0 = ldA16(&XS[rbase][0], c0);
        const v16bf a1 = ldA16(&XS[rbase + 16][0], c0);
#pragma unroll
        for (int tn = 0; tn < 4; ++tn) {
            const v16bf bf = ldB16(&WS[tn * 16 + (l & 15)][(l >> 4) * 16]);
            acc[0][tn] = wmma_bf16(a0, bf, acc[0][tn]);
            acc[1][tn] = wmma_bf16(a1, bf, acc[1][tn]);
        }
        __syncthreads();
    }

    // epilogue: C/D layout -> head-major scratch
#pragma unroll
    for (int rf = 0; rf < 2; ++rf) {
#pragma unroll
        for (int tn = 0; tn < 4; ++tn) {
#pragma unroll
            for (int j = 0; j < 8; ++j) {
                const int n   = nbase + tn * 16 + (l & 15);
                const int row = mbase + wv * 32 + rf * 16 + ((l < 16) ? j : j + 8);
                float val = acc[rf][tn][j] + bias[n];
                if (MODE == 0) {
                    val *= 0.125f;  // fold 1/sqrt(Dh) into q
                    const int h = n >> 6, d = n & 63;
                    q_s[((size_t)h * TLEN + row) * DH + d] = (bf16_t)val;
                } else {
                    const int b = row >> 13, kk = row & (KLEN - 1);
                    if (n < CDIM) {
                        const int h = n >> 6, d = n & 63;
                        k_s[(((size_t)(b * HEADS + h)) * KLEN + kk) * DH + d] = (bf16_t)val;
                    } else {
                        const int n2 = n - CDIM;
                        const int h = n2 >> 6, d = n2 & 63;
                        v_t[(((size_t)(b * HEADS + h)) * DH + d) * KLEN + kk] = (bf16_t)val;
                    }
                }
            }
        }
    }
}

// ---------------------------------------------------------------------------
// Kernel 2: flash attention per (b,h). Grid = B*H*2 WGs; each of 8 waves owns
// a 16-query strip; 64-key blocks double-buffered via async-to-LDS; online
// softmax (max via ds_swizzle, sum via P@ones WMMA); WMMA S and ctx.
// ---------------------------------------------------------------------------
__global__ __launch_bounds__(256) void attn_kernel(
    const bf16_t* __restrict__ q_s, const bf16_t* __restrict__ k_s,
    const bf16_t* __restrict__ v_t, bf16_t* __restrict__ ctx_s)
{
    __shared__ bf16_t kS[2][64][72];     // keys x dh   (+8 pad)
    __shared__ bf16_t vS[2][64][72];     // dh   x keys (+8 pad)
    __shared__ bf16_t pS[8][16][72];     // per-wave P relayout buffer

    const int bx = blockIdx.x;
    const int tchunk = bx & 1;
    const int bh = bx >> 1;
    const int h = bh & (HEADS - 1);
    const int b = bh >> 3;

    const int tid = threadIdx.x;
    const int wv  = tid >> 5;
    const int l   = tid & 31;
    const int qbase = tchunk * 128 + wv * 16;
    const int c0 = (l < 16) ? 0 : 8;

    // q A-fragments for this wave's 16 rows (two 32-wide slices of Dh=64)
    const bf16_t* qrow = q_s + ((size_t)h * TLEN + qbase + (l & 15)) * DH;
    const v16bf qf0 = ldA16(qrow, c0);
    const v16bf qf1 = ldA16(qrow + 32, c0);

    v16bf ones;
#pragma unroll
    for (int i = 0; i < 16; ++i) ones[i] = (bf16_t)1.0f;

    v8f cacc[4] = {};
    float m[8], lsum[8];
#pragma unroll
    for (int j = 0; j < 8; ++j) { m[j] = -3.0e38f; lsum[j] = 0.0f; }

    const size_t kbase = ((size_t)(b * HEADS + h)) * KLEN * DH;
    const size_t vbase = ((size_t)(b * HEADS + h)) * DH * KLEN;

    // --- staging: 64 keys x 64 dh of k and v^T (16 KB), 64B per thread ------
#if HAVE_ASYNC_LDS
    auto stage_issue = [&](int buf, int kb) {
        if (tid < 128) {
            const int key = tid >> 1, dof = (tid & 1) * 32;
            const bf16_t* src = k_s + kbase + (size_t)(kb + key) * DH + dof;
#pragma unroll
            for (int c = 0; c < 4; ++c)
                async_b128(src + 8 * c, &kS[buf][key][dof + 8 * c]);
        } else {
            const int t2 = tid - 128;
            const int dr = t2 >> 1, kof = (t2 & 1) * 32;
            const bf16_t* src = v_t + vbase + (size_t)dr * KLEN + kb + kof;
#pragma unroll
            for (int c = 0; c < 4; ++c)
                async_b128(src + 8 * c, &vS[buf][dr][kof + 8 * c]);
        }
    };
#else
    auto stage_sync = [&](int buf, int kb) {
        if (tid < 128) {
            const int key = tid >> 1, dof = (tid & 1) * 32;
            const uint4* src = reinterpret_cast<const uint4*>(
                k_s + kbase + (size_t)(kb + key) * DH + dof);
#pragma unroll
            for (int c = 0; c < 4; ++c)
                *reinterpret_cast<uint4*>(&kS[buf][key][dof + 8 * c]) = src[c];
        } else {
            const int t2 = tid - 128;
            const int dr = t2 >> 1, kof = (t2 & 1) * 32;
            const uint4* src = reinterpret_cast<const uint4*>(
                v_t + vbase + (size_t)dr * KLEN + kb + kof);
#pragma unroll
            for (int c = 0; c < 4; ++c)
                *reinterpret_cast<uint4*>(&vS[buf][dr][kof + 8 * c]) = src[c];
        }
    };
#endif

    auto compute = [&](int buf) {
        // S = q @ k^T : four 16x16 key tiles, Dh reduced in 2 WMMA steps each
        v8f s[4];
#pragma unroll
        for (int tn = 0; tn < 4; ++tn) {
            v8f z = {};
            const bf16_t* kp = &kS[buf][tn * 16 + (l & 15)][(l >> 4) * 16];
            z = wmma_bf16(qf0, ldB16(kp), z);
            z = wmma_bf16(qf1, ldB16(kp + 32), z);
            s[tn] = z;
        }
        // online softmax: row j (+8 upper half) spans a 16-lane half
        float al[8];
#pragma unroll
        for (int j = 0; j < 8; ++j) {
            float mx = fmaxf(fmaxf(s[0][j], s[1][j]), fmaxf(s[2][j], s[3][j]));
            mx = max_red16(mx);
            const float mn = fmaxf(m[j], mx);
            al[j] = __expf(m[j] - mn);
            m[j] = mn;
            const int prow = (l < 16) ? j : j + 8;
#pragma unroll
            for (int tn = 0; tn < 4; ++tn)
                pS[wv][prow][tn * 16 + (l & 15)] = (bf16_t)__expf(s[tn][j] - mn);
#pragma unroll
            for (int tn = 0; tn < 4; ++tn) cacc[tn][j] *= al[j];
        }
        // reload P as A-fragments (keys 0..31, 32..63)
        const v16bf pf0 = ldA16(&pS[wv][(l & 15)][0],  c0);
        const v16bf pf1 = ldA16(&pS[wv][(l & 15)][32], c0);
        // row sums via P @ ones (replaces shuffle-reduce)
        v8f ss = {};
        ss = wmma_bf16(pf0, ones, ss);
        ss = wmma_bf16(pf1, ones, ss);
#pragma unroll
        for (int j = 0; j < 8; ++j) lsum[j] = lsum[j] * al[j] + ss[j];
        // ctx += P @ V
#pragma unroll
        for (int tn = 0; tn < 4; ++tn) {
            const bf16_t* vp = &vS[buf][tn * 16 + (l & 15)][(l >> 4) * 16];
            cacc[tn] = wmma_bf16(pf0, ldB16(vp), cacc[tn]);
            cacc[tn] = wmma_bf16(pf1, ldB16(vp + 32), cacc[tn]);
        }
    };

    const int nIter = KLEN / 64;
#if HAVE_ASYNC_LDS
    stage_issue(0, 0);
    for (int it = 0; it < nIter; ++it) {
        const int cur = it & 1;
        WAIT_ASYNC();            // own async stores to LDS done
        __syncthreads();         // everyone's data visible; prev reads done
        if (it + 1 < nIter) stage_issue(1 - cur, (it + 1) * 64);
        compute(cur);
    }
#else
    for (int it = 0; it < nIter; ++it) {
        stage_sync(0, it * 64);
        __syncthreads();
        compute(0);
        __syncthreads();
    }
#endif

    // epilogue: ctx/lsum -> ctx_s (B*T, C) bf16
#pragma unroll
    for (int j = 0; j < 8; ++j) {
        const float inv = 1.0f / lsum[j];
        const int trow = qbase + ((l < 16) ? j : j + 8);
#pragma unroll
        for (int tn = 0; tn < 4; ++tn) {
            const int dh = tn * 16 + (l & 15);
            ctx_s[((size_t)(b * TLEN + trow)) * CDIM + h * DH + dh] =
                (bf16_t)(cacc[tn][j] * inv);
        }
    }
}

// ---------------------------------------------------------------------------
// Kernel 3: out-proj GEMM  y = ctx(bf16) @ Wo^T + bo + queries  (f32 out)
// 256x64 tiling; A already bf16, W staged f32->bf16.
// ---------------------------------------------------------------------------
__global__ __launch_bounds__(256) void outproj_kernel(
    const bf16_t* __restrict__ ctx_s, const float* __restrict__ Wo,
    const float* __restrict__ bo, const float* __restrict__ qt,
    float* __restrict__ y)
{
    __shared__ bf16_t XS[256][40];
    __shared__ bf16_t WS[64][40];

    const int tid   = threadIdx.x;
    const int wv    = tid >> 5;
    const int l     = tid & 31;
    const int nbase = blockIdx.x * 64;
    const int mbase = blockIdx.y * 256;

    v8f acc[2][4] = {};

    for (int kc = 0; kc < CDIM; kc += 32) {
        {   // A: copy bf16 tile (256x32), one row (64B) per thread
            const uint4* src = reinterpret_cast<const uint4*>(
                ctx_s + (size_t)(mbase + tid) * CDIM + kc);
#pragma unroll
            for (int c = 0; c < 4; ++c)
                *reinterpret_cast<uint4*>(&XS[tid][8 * c]) = src[c];
        }
#pragma unroll
        for (int p = 0; p < 2; ++p) {
            const int r = (tid >> 3) + 32 * p;
            const int c = (tid & 7) << 2;
            const float4 f = *reinterpret_cast<const float4*>(
                Wo + (size_t)(nbase + r) * CDIM + kc + c);
            WS[r][c + 0] = (bf16_t)f.x; WS[r][c + 1] = (bf16_t)f.y;
            WS[r][c + 2] = (bf16_t)f.z; WS[r][c + 3] = (bf16_t)f.w;
        }
        __syncthreads();

        const int rbase = wv * 32 + (l & 15);
        const int c0    = (l < 16) ? 0 : 8;
        const v16bf a0 = ldA16(&XS[rbase][0], c0);
        const v16bf a1 = ldA16(&XS[rbase + 16][0], c0);
#pragma unroll
        for (int tn = 0; tn < 4; ++tn) {
            const v16bf bf = ldB16(&WS[tn * 16 + (l & 15)][(l >> 4) * 16]);
            acc[0][tn] = wmma_bf16(a0, bf, acc[0][tn]);
            acc[1][tn] = wmma_bf16(a1, bf, acc[1][tn]);
        }
        __syncthreads();
    }

#pragma unroll
    for (int rf = 0; rf < 2; ++rf) {
#pragma unroll
        for (int tn = 0; tn < 4; ++tn) {
#pragma unroll
            for (int j = 0; j < 8; ++j) {
                const int n   = nbase + tn * 16 + (l & 15);
                const int row = mbase + wv * 32 + rf * 16 + ((l < 16) ? j : j + 8);
                y[(size_t)row * CDIM + n] =
                    acc[rf][tn][j] + bo[n] + qt[(size_t)(row & (TLEN - 1)) * CDIM + n];
            }
        }
    }
}

// ---------------------------------------------------------------------------
// Kernel 4: LayerNorm over C=512 per row (2048 rows)
// ---------------------------------------------------------------------------
__global__ __launch_bounds__(256) void ln_kernel(
    const float* __restrict__ y, const float* __restrict__ gamma,
    const float* __restrict__ beta, float* __restrict__ out)
{
    __shared__ float s1[256], s2[256];
    const int row = blockIdx.x;
    const int tid = threadIdx.x;
    const float v0 = y[(size_t)row * CDIM + tid];
    const float v1 = y[(size_t)row * CDIM + 256 + tid];
    s1[tid] = v0 + v1;
    s2[tid] = v0 * v0 + v1 * v1;
    __syncthreads();
    for (int o = 128; o > 0; o >>= 1) {
        if (tid < o) { s1[tid] += s1[tid + o]; s2[tid] += s2[tid + o]; }
        __syncthreads();
    }
    const float mu  = s1[0] * (1.0f / CDIM);
    const float var = s2[0] * (1.0f / CDIM) - mu * mu;
    const float r   = rsqrtf(var + 1e-5f);
    out[(size_t)row * CDIM + tid]       = (v0 - mu) * r * gamma[tid] + beta[tid];
    out[(size_t)row * CDIM + 256 + tid] = (v1 - mu) * r * gamma[256 + tid] + beta[256 + tid];
}

// ---------------------------------------------------------------------------
extern "C" void kernel_launch(void* const* d_in, const int* in_sizes, int n_in,
                              void* d_out, int out_size, void* d_ws, size_t ws_size,
                              hipStream_t stream)
{
    (void)in_sizes; (void)n_in; (void)out_size; (void)ws_size;
    const float* x   = (const float*)d_in[0];   // (B,K,C)
    const float* qt  = (const float*)d_in[1];   // (1,T,C)
    const float* wIn = (const float*)d_in[2];   // (3C,C)
    const float* bIn = (const float*)d_in[3];   // (3C)
    const float* wO  = (const float*)d_in[4];   // (C,C)
    const float* bO  = (const float*)d_in[5];   // (C)
    const float* g   = (const float*)d_in[6];   // (C)
    const float* be  = (const float*)d_in[7];   // (C)
    float* out = (float*)d_out;

    char* ws = (char*)d_ws;
    bf16_t* q_s   = (bf16_t*)(ws);                                                   // 256 KiB
    bf16_t* k_s   = (bf16_t*)(ws + (size_t)262144);                                  // 64 MiB
    bf16_t* v_t   = (bf16_t*)(ws + (size_t)262144 + (size_t)67108864);               // 64 MiB
    bf16_t* ctx_s = (bf16_t*)(ws + (size_t)262144 + 2 * (size_t)67108864);           // 2 MiB
    float*  y     = (float*) (ws + (size_t)262144 + 2 * (size_t)67108864 + 2097152); // 4 MiB

    // q projection: M=256, N=512
    proj_kernel<0><<<dim3(8, 1), 256, 0, stream>>>(qt, wIn, bIn, q_s, nullptr, nullptr);
    // k,v projection: M=65536, N=1024 (W rows 512..1535)
    proj_kernel<1><<<dim3(16, 256), 256, 0, stream>>>(
        x, wIn + (size_t)CDIM * CDIM, bIn + CDIM, nullptr, k_s, v_t);
    // flash attention: B*H*2 workgroups
    attn_kernel<<<dim3(BATCH * HEADS * 2), 256, 0, stream>>>(q_s, k_s, v_t, ctx_s);
    // out projection + residual: M=2048, N=512
    outproj_kernel<<<dim3(8, 8), 256, 0, stream>>>(ctx_s, wO, bO, qt, y);
    // layernorm
    ln_kernel<<<dim3(BATCH * TLEN), 256, 0, stream>>>(y, g, be, out);
}
